// AscendQwen3Next_GatedDeltaNet_16338055594626
// MI455X (gfx1250) — compile-verified, matrix operand order checked by reading, not run on hardware
//
#include <hip/hip_runtime.h>
#include <hip/hip_bf16.h>

// ---------------------------------------------------------------------------
// Gated DeltaNet (Qwen3-Next style) forward for MI455X (gfx1250, wave32).
//  * GEMMs: v_wmma_f32_16x16x32_f16, f32 accumulate, A/B pre-converted to f16
//    (weights pre-transposed to K-major so both tiles stage identically).
//  * Tile staging: global_load_async_to_lds_b128 + s_wait_asynccnt, LDS
//    double-buffered (2 x 16KB of the 320KB WGP pool).
//  * Delta-rule scan: 128x128 f32 state register-resident, 1 block/head.
// ---------------------------------------------------------------------------

#define L_   4096
#define H_   2048
#define HK_  16
#define HV_  32
#define DK_  128
#define DV_  128
#define KW_  4
#define QKVZ_ 12288   // HK * (2*DK + 2*R*DV)
#define BA_   64      // HK * 2*R
#define BA_PAD_ 128   // padded N for the ba GEMM (async staging never OOB)
#define CONVD_ 8192   // 2*HK*DK + HV*DV
#define EPS_  1e-6f
#define INV_SQRT_DK 0.08838834764831845f

typedef __attribute__((ext_vector_type(16))) _Float16 v16h;
typedef __attribute__((ext_vector_type(8)))  float    v8f;
typedef __attribute__((ext_vector_type(8)))  _Float16 h8;

// ---------------------------------------------------------------------------
// gfx1250 async global -> LDS, 16B per lane (one wave = 512B).
// Low 32 bits of a generic pointer to __shared__ are the LDS byte offset.
// ---------------------------------------------------------------------------
__device__ __forceinline__ void async_b128(const void* gaddr, void* laddr)
{
    const unsigned lds_off = (unsigned)(uintptr_t)laddr;
    asm volatile("global_load_async_to_lds_b128 %0, %1, off"
                 :: "v"(lds_off), "v"(gaddr)
                 : "memory");
}

// ---------------------------------------------------------------------------
// C[M,N] = A[M,K] * Bt[N,K]^T  (both f16, K-major; f32 accumulate/output).
// 256 threads = 8 waves; macro-tile 128(M) x 128(N); K in 32-slices,
// double-buffered LDS staged entirely with async copies.
// ---------------------------------------------------------------------------
__device__ __forceinline__ void stage_tile(const _Float16* gA, const _Float16* gB,
                                           _Float16 (*As)[32], _Float16 (*Bs)[32],
                                           int tid, int K)
{
    #pragma unroll
    for (int i = 0; i < 2; ++i) {
        const int lin  = tid * 2 + i;     // 0..511 : 128 rows x 4 x 16B
        const int m    = lin >> 2;
        const int part = (lin & 3) * 8;   // halves
        async_b128(gA + (size_t)m * K + part, &As[m][part]);
        async_b128(gB + (size_t)m * K + part, &Bs[m][part]);
    }
}

__global__ __launch_bounds__(256)
void gdn_gemm_wmma_f16(const _Float16* __restrict__ A, const _Float16* __restrict__ Bt,
                       float* __restrict__ C, int M, int N, int K)
{
    __shared__ alignas(16) _Float16 As[2][128][32];   // [buf][m][k]
    __shared__ alignas(16) _Float16 Bs[2][128][32];   // [buf][n][k]

    const int tid  = threadIdx.x;
    const int lane = tid & 31;
    const int wave = tid >> 5;
    const int wm   = wave >> 1;          // 0..3
    const int wn   = wave & 1;           // 0..1
    const int bm   = blockIdx.y * 128;
    const int bn   = blockIdx.x * 128;

    v8f acc[2][4] = {};

    const _Float16* gA = A  + (size_t)bm * K;
    const _Float16* gB = Bt + (size_t)bn * K;

    const int nk = K >> 5;
    stage_tile(gA, gB, As[0], Bs[0], tid, K);

    for (int kt = 0; kt < nk; ++kt) {
        const int cur = kt & 1;
        if (kt + 1 < nk) {
            stage_tile(gA + (kt + 1) * 32, gB + (kt + 1) * 32,
                       As[cur ^ 1], Bs[cur ^ 1], tid, K);
            // 4 wave-transfers of the NEXT tile stay in flight; wait for
            // the 4 belonging to the CURRENT tile (in-order completion).
            asm volatile("s_wait_asynccnt 0x4" ::: "memory");
        } else {
            asm volatile("s_wait_asynccnt 0x0" ::: "memory");
        }
        __syncthreads();

        // ISA 16-bit A layout: lane l -> M = l&15, K = (l>>4)*8 + {0..7,16..23}
        const int l15 = lane & 15;
        const int ks  = (lane >> 4) * 8;

        v16h afr[2];
        #pragma unroll
        for (int i = 0; i < 2; ++i) {
            const int mr = wm * 32 + i * 16 + l15;
            h8 lo = *(const h8*)&As[cur][mr][ks];
            h8 hi = *(const h8*)&As[cur][mr][ks + 16];
            #pragma unroll
            for (int e = 0; e < 8; ++e) { afr[i][e] = lo[e]; afr[i][8 + e] = hi[e]; }
        }
        v16h bfr[4];
        #pragma unroll
        for (int jf = 0; jf < 4; ++jf) {
            const int nr = wn * 64 + jf * 16 + l15;
            h8 lo = *(const h8*)&Bs[cur][nr][ks];
            h8 hi = *(const h8*)&Bs[cur][nr][ks + 16];
            #pragma unroll
            for (int e = 0; e < 8; ++e) { bfr[jf][e] = lo[e]; bfr[jf][8 + e] = hi[e]; }
        }

        #pragma unroll
        for (int i = 0; i < 2; ++i)
            #pragma unroll
            for (int jf = 0; jf < 4; ++jf)
                acc[i][jf] = __builtin_amdgcn_wmma_f32_16x16x32_f16(
                    false, afr[i], false, bfr[jf], (short)0, acc[i][jf],
                    false, false);
        __syncthreads();
    }

    // store: C/D layout: lane l -> N = l&15, VGPR r -> M = r + (l>>4)*8
    const int l15 = lane & 15;
    const int mhl = (lane >> 4) * 8;
    #pragma unroll
    for (int i = 0; i < 2; ++i) {
        #pragma unroll
        for (int jf = 0; jf < 4; ++jf) {
            const int n = bn + wn * 64 + jf * 16 + l15;
            if (n < N) {
                #pragma unroll
                for (int r = 0; r < 8; ++r) {
                    const int m = bm + wm * 32 + i * 16 + mhl + r;
                    C[(size_t)m * N + n] = acc[i][jf][r];
                }
            }
        }
    }
}

// ---------------------------------------------------------------------------
// One-time precision/layout prep (all streams fit in the 192MB L2 afterwards)
// ---------------------------------------------------------------------------
__global__ __launch_bounds__(256)
void gdn_f32_to_f16(const float* __restrict__ x, _Float16* __restrict__ y, size_t n)
{
    const size_t i = (size_t)blockIdx.x * 256 + threadIdx.x;
    if (i < n) y[i] = (_Float16)x[i];
}

// src[K][N] (f32) -> dst[Np][K] (f16), zero-padded rows N..Np-1
__global__ __launch_bounds__(256)
void gdn_transpose_f16(const float* __restrict__ src, _Float16* __restrict__ dst,
                       int K, int N, int Np)
{
    __shared__ float s[32][33];
    const int tid = threadIdx.x;
    const int tx = tid & 31, ty = tid >> 5;
    const int n0 = blockIdx.x * 32, k0 = blockIdx.y * 32;
    #pragma unroll
    for (int r = 0; r < 4; ++r) {
        const int k = k0 + ty + r * 8, n = n0 + tx;
        s[ty + r * 8][tx] = (n < N) ? src[(size_t)k * N + n] : 0.0f;
    }
    __syncthreads();
    #pragma unroll
    for (int r = 0; r < 4; ++r) {
        const int n = n0 + ty + r * 8, k = k0 + tx;
        if (n < Np) dst[(size_t)n * K + k] = (_Float16)s[tx][ty + r * 8];
    }
}

// ---------------------------------------------------------------------------
// Causal depthwise conv (KW=4) + bias + SiLU. Output col c maps into qkvz.
// mixed = [ q(n,HK*DK) | k(n,HK*DK) | v(n,HV*DV) ]
// ---------------------------------------------------------------------------
__global__ __launch_bounds__(256)
void gdn_conv_silu(const float* __restrict__ qkvz, const float* __restrict__ cw,
                   const float* __restrict__ cb, float* __restrict__ conv)
{
    const size_t idx = (size_t)blockIdx.x * blockDim.x + threadIdx.x;
    if (idx >= (size_t)L_ * CONVD_) return;
    const int t = (int)(idx >> 13);
    const int c = (int)(idx & 8191);

    int qcol;
    if (c < 2048)      { int h = c >> 7;            qcol = h * 768 + (c & 127); }
    else if (c < 4096) { int cc = c - 2048; int h = cc >> 7;
                         qcol = h * 768 + 128 + (cc & 127); }
    else               { int cc = c - 4096; int vh = cc >> 7;
                         qcol = (vh >> 1) * 768 + 256 + (vh & 1) * 128 + (cc & 127); }

    float acc = cb[c];
    #pragma unroll
    for (int j = 0; j < KW_; ++j) {
        const int tt = t - (KW_ - 1) + j;
        if (tt >= 0)
            acc += qkvz[(size_t)tt * QKVZ_ + qcol] * cw[c * KW_ + j];
    }
    conv[idx] = acc / (1.0f + expf(-acc));     // silu
}

// ---------------------------------------------------------------------------
__device__ __forceinline__ float block_reduce_sum_128(float v, float* red, int tid)
{
    red[tid] = v; __syncthreads();
    #pragma unroll
    for (int s = 64; s > 0; s >>= 1) {
        if (tid < s) red[tid] += red[tid + s];
        __syncthreads();
    }
    float r = red[0]; __syncthreads();
    return r;
}

// In-place L2 normalization of q (segments 0..15, *DK^-0.5) and k (16..31).
__global__ __launch_bounds__(128)
void gdn_norm_qk(float* __restrict__ conv)
{
    __shared__ float red[128];
    const int gid = blockIdx.x;
    const int t   = gid >> 5;
    const int seg = gid & 31;
    const int tid = threadIdx.x;
    const size_t base = (size_t)t * CONVD_ + (size_t)seg * 128;
    float x = conv[base + tid];
    float ss = block_reduce_sum_128(x * x, red, tid);
    float scale = rsqrtf(ss + EPS_) * (seg < 16 ? INV_SQRT_DK : 1.0f);
    conv[base + tid] = x * scale;
}

// g = -exp(A_log)*softplus(a + dt_bias); beta = sigmoid(b)
__global__ __launch_bounds__(256)
void gdn_gbeta(const float* __restrict__ ba, const float* __restrict__ A_log,
               const float* __restrict__ dt_bias,
               float* __restrict__ g, float* __restrict__ beta)
{
    const int idx = blockIdx.x * blockDim.x + threadIdx.x;
    if (idx >= L_ * HV_) return;
    const int t = idx >> 5, hv = idx & 31;
    const int h = hv >> 1, r = hv & 1;
    const float bv = ba[(size_t)t * BA_ + h * 4 + r];
    const float av = ba[(size_t)t * BA_ + h * 4 + 2 + r];
    const float x  = av + dt_bias[hv];
    const float sp = (x > 20.0f) ? x : log1pf(expf(x));
    g[idx]    = -expf(A_log[hv]) * sp;
    beta[idx] = 1.0f / (1.0f + expf(-bv));
}

// ---------------------------------------------------------------------------
// Gated delta-rule scan. One block per value head; thread j owns state
// column S[:, j] (128 f32 registers). k/q broadcast through LDS per step.
// ---------------------------------------------------------------------------
__global__ __launch_bounds__(128)
void gdn_scan(const float* __restrict__ conv, const float* __restrict__ g,
              const float* __restrict__ beta, float* __restrict__ o)
{
    const int vh = blockIdx.x;          // 0..31
    const int kh = vh >> 1;             // key head
    const int j  = threadIdx.x;         // dv column

    __shared__ float ks_[128];
    __shared__ float qs_[128];
    __shared__ float sc_[2];

    float S[DK_];
    #pragma unroll
    for (int d = 0; d < DK_; ++d) S[d] = 0.0f;

    for (int t = 0; t < L_; ++t) {
        const size_t row = (size_t)t * CONVD_;
        qs_[j] = conv[row + kh * 128 + j];
        ks_[j] = conv[row + 2048 + kh * 128 + j];
        const float vv = conv[row + 4096 + vh * 128 + j];
        if (j == 0) {
            sc_[0] = expf(g[t * HV_ + vh]);
            sc_[1] = beta[t * HV_ + vh];
        }
        __syncthreads();

        const float eg = sc_[0], bt = sc_[1];
        float kv = 0.0f;
        #pragma unroll
        for (int d = 0; d < DK_; ++d) { S[d] *= eg; kv += ks_[d] * S[d]; }
        const float delta = (vv - kv) * bt;
        float oacc = 0.0f;
        #pragma unroll
        for (int d = 0; d < DK_; ++d) { S[d] += ks_[d] * delta; oacc += qs_[d] * S[d]; }

        o[(size_t)t * (HV_ * DV_) + vh * 128 + j] = oacc;
        __syncthreads();
    }
}

// o16 = f16( RMSNorm(o * silu(z)) * norm_w ) ; z read straight out of qkvz
__global__ __launch_bounds__(128)
void gdn_gate_norm(const float* __restrict__ qkvz, const float* __restrict__ norm_w,
                   const float* __restrict__ o, _Float16* __restrict__ o16)
{
    __shared__ float red[128];
    const int gid = blockIdx.x;
    const int t   = gid >> 5;
    const int vh  = gid & 31;
    const int tid = threadIdx.x;
    const int h = vh >> 1, r = vh & 1;

    const float z = qkvz[(size_t)t * QKVZ_ + h * 768 + 512 + r * 128 + tid];
    const size_t oidx = (size_t)t * (HV_ * DV_) + vh * 128 + tid;
    float val = o[oidx] * (z / (1.0f + expf(-z)));
    float ms  = block_reduce_sum_128(val * val, red, tid) * (1.0f / 128.0f);
    o16[oidx] = (_Float16)(val * rsqrtf(ms + EPS_) * norm_w[tid]);
}

// ---------------------------------------------------------------------------
extern "C" void kernel_launch(void* const* d_in, const int* in_sizes, int n_in,
                              void* d_out, int out_size, void* d_ws, size_t ws_size,
                              hipStream_t stream)
{
    const float* hidden  = (const float*)d_in[0];
    const float* W_qkvz  = (const float*)d_in[1];
    const float* W_ba    = (const float*)d_in[2];
    const float* conv_w  = (const float*)d_in[3];
    const float* conv_b  = (const float*)d_in[4];
    const float* A_log   = (const float*)d_in[5];
    const float* dt_bias = (const float*)d_in[6];
    const float* norm_w  = (const float*)d_in[7];
    const float* W_out   = (const float*)d_in[8];
    float* out = (float*)d_out;

    // workspace layout
    float* ws   = (float*)d_ws;
    float* qkvz = ws;                                   // L * 12288
    float* ba   = qkvz + (size_t)L_ * QKVZ_;            // L * 64
    float* conv = ba   + (size_t)L_ * BA_;              // L * 8192
    float* gbuf = conv + (size_t)L_ * CONVD_;           // L * 32
    float* bbuf = gbuf + (size_t)L_ * HV_;              // L * 32
    float* obuf = bbuf + (size_t)L_ * HV_;              // L * 4096

    _Float16* hidden16 = (_Float16*)(obuf + (size_t)L_ * HV_ * DV_);
    _Float16* Wqkvz_t  = hidden16 + (size_t)L_ * H_;            // [12288][2048]
    _Float16* Wba_t    = Wqkvz_t  + (size_t)QKVZ_ * H_;         // [128][2048]
    _Float16* Wout_t   = Wba_t    + (size_t)BA_PAD_ * H_;       // [2048][4096]
    _Float16* o16      = Wout_t   + (size_t)H_ * (HV_ * DV_);   // [4096][4096]

    // 0) precision/layout prep (one-time; thereafter weights live in L2)
    gdn_f32_to_f16<<<dim3((L_ * H_ + 255) / 256), 256, 0, stream>>>(
        hidden, hidden16, (size_t)L_ * H_);
    gdn_transpose_f16<<<dim3(QKVZ_ / 32, H_ / 32), 256, 0, stream>>>(
        W_qkvz, Wqkvz_t, H_, QKVZ_, QKVZ_);
    gdn_transpose_f16<<<dim3(BA_PAD_ / 32, H_ / 32), 256, 0, stream>>>(
        W_ba, Wba_t, H_, BA_, BA_PAD_);
    gdn_transpose_f16<<<dim3(H_ / 32, (HV_ * DV_) / 32), 256, 0, stream>>>(
        W_out, Wout_t, HV_ * DV_, H_, H_);

    // 1) qkvz = hidden @ W_qkvz   (4096 x 2048 x 12288)
    gdn_gemm_wmma_f16<<<dim3(QKVZ_ / 128, L_ / 128), 256, 0, stream>>>(
        hidden16, Wqkvz_t, qkvz, L_, QKVZ_, H_);

    // 2) ba = hidden @ W_ba       (4096 x 2048 x 64, N padded to 128)
    gdn_gemm_wmma_f16<<<dim3(1, L_ / 128), 256, 0, stream>>>(
        hidden16, Wba_t, ba, L_, BA_, H_);

    // 3) causal conv + silu
    {
        const size_t tot = (size_t)L_ * CONVD_;
        gdn_conv_silu<<<dim3((unsigned)((tot + 255) / 256)), 256, 0, stream>>>(
            qkvz, conv_w, conv_b, conv);
    }

    // 4) l2-normalize q,k in place
    gdn_norm_qk<<<dim3(L_ * 32), 128, 0, stream>>>(conv);

    // 5) gate scalars
    gdn_gbeta<<<dim3((L_ * HV_ + 255) / 256), 256, 0, stream>>>(
        ba, A_log, dt_bias, gbuf, bbuf);

    // 6) sequential delta-rule scan, one block per value head
    gdn_scan<<<dim3(HV_), 128, 0, stream>>>(conv, gbuf, bbuf, obuf);

    // 7) silu(z) gate + RMSNorm -> f16
    gdn_gate_norm<<<dim3(L_ * HV_), 128, 0, stream>>>(qkvz, norm_w, obuf, o16);

    // 8) out = o @ W_out          (4096 x 4096 x 2048)
    gdn_gemm_wmma_f16<<<dim3(H_ / 128, L_ / 128), 256, 0, stream>>>(
        o16, Wout_t, out, L_, H_, HV_ * DV_);
}
